// SelfAttention_6622839570725
// MI455X (gfx1250) — compile-verified, matrix operand order checked by reading, not run on hardware
//
#include <hip/hip_runtime.h>
#include <hip/hip_bf16.h>

// Self-attention (causal, 16 heads, HD=64) for MI455X gfx1250.
// Pipeline: f32->f16 convert / weight transpose -> WMMA QKV GEMM (async
// LDS-staged B panel, double-buffered, software-pipelined A, fence-free
// split barriers) -> flash attention (online softmax, WMMA for QK^T and PV,
// latency-hoisted V loads) -> WMMA proj GEMM. Workspace usage: ~88 MB.

#define Bq 4
#define Tq 2048
#define Eq 1024
#define Hq 16
#define HDq 64

typedef _Float16 v8h  __attribute__((ext_vector_type(8)));
typedef _Float16 v16h __attribute__((ext_vector_type(16)));
typedef float    v8f  __attribute__((ext_vector_type(8)));

#define NEG_INF (-__builtin_huge_valf())

// Load one WMMA fragment: 16 halfs per lane as two 16-byte loads.
// Caller passes p = row_base + klo (klo = (lane>>4)*8). Elements map to
// K = {klo..klo+7, klo+16..klo+23} per the CDNA5 16-bit A-matrix layout.
__device__ __forceinline__ v16h load_frag(const _Float16* __restrict__ p) {
  v8h lo = *(const v8h*)(p);
  v8h hi = *(const v8h*)(p + 16);
  v16h r;
#pragma unroll
  for (int i = 0; i < 8; ++i) { r[i] = lo[i]; r[i + 8] = hi[i]; }
  return r;
}

__device__ __forceinline__ v8f wmma_f16(v16h a, v16h b, v8f c) {
  return __builtin_amdgcn_wmma_f32_16x16x32_f16(false, a, false, b, (short)0, c,
                                                false, false);
}

// Wait for this wave's async global->LDS staging, then cross-wave split
// barrier WITHOUT the full memory fence __syncthreads() implies (no
// s_wait_loadcnt 0: in-flight A prefetch loads target registers only and
// need no cross-wave ordering; ds reads were already drained by the dscnt
// waits in front of the WMMAs that consumed them).
__device__ __forceinline__ void async_stage_barrier() {
  asm volatile(
      "s_wait_asynccnt 0\n\t"
      "s_barrier_signal -1\n\t"
      "s_barrier_wait -1" ::: "memory");
}

// ---------------------------------------------------------------- converts ---
__global__ void f32_to_f16_kernel(const float* __restrict__ src,
                                  _Float16* __restrict__ dst, int n) {
  int i = blockIdx.x * blockDim.x + threadIdx.x;
  if (i < n) dst[i] = (_Float16)src[i];
}

// W: [K][N] f32 row-major  ->  Wt: [N][K] f16 row-major
__global__ void transpose_to_f16_kernel(const float* __restrict__ W,
                                        _Float16* __restrict__ Wt, int K, int N) {
  int i = blockIdx.x * blockDim.x + threadIdx.x;
  if (i < N * K) {
    int n = i / K, k = i - n * K;
    Wt[i] = (_Float16)W[(size_t)k * N + n];
  }
}

// ----------------------------------------------------- shared GEMM mainloop --
// Block = 8 waves, block tile 256x64: wave w owns rows m0w..m0w+31 (two
// 16-row WMMA tiles), all waves share the 64-column B panel staged in LDS by
// the async global->LDS engine (double-buffered). A fragments are software-
// pipelined one k-step ahead; the four B fragments feed 8 WMMAs per k-step.
__device__ __forceinline__ void gemm_tile_256x64(
    const _Float16* __restrict__ Ah, const _Float16* __restrict__ Bt,
    _Float16 (*bsh)[64 * 32], int m0w, int n0, v8f acc[2][4]) {
  const int K = Eq;
  const int tid = threadIdx.x;
  const int lane = tid & 31;
  const int r = lane & 15, klo = (lane >> 4) << 3;

  // Async staging role: thread tid loads 16 bytes: column n = tid/4,
  // k-chunk = (tid%4)*8 halfs. 256 threads cover the 64x32 f16 panel exactly.
  const int nrow = tid >> 2, chunk = tid & 3;
  const unsigned gbase = (unsigned)((((n0 + nrow) * K) + chunk * 8) * 2);
  const unsigned lb0 = (unsigned)(uintptr_t)&bsh[0][nrow * 32 + chunk * 8];
  const unsigned lb1 = (unsigned)(uintptr_t)&bsh[1][nrow * 32 + chunk * 8];

  // Prologue: stage k-slice 0 into buffer 0; prefetch first A fragments.
  asm volatile("global_load_async_to_lds_b128 %0, %1, %2"
               :: "v"(lb0), "v"(gbase), "s"(Bt) : "memory");
  const _Float16* arow0 = Ah + (size_t)(m0w + r) * K + klo;
  const _Float16* arow1 = arow0 + (size_t)16 * K;
  v16h a0 = load_frag(arow0);
  v16h a1 = load_frag(arow1);
  async_stage_barrier();

  for (int kb = 0; kb < K; kb += 32) {
    const int cur = (kb >> 5) & 1;
    const int kn = (kb + 32 < K) ? kb + 32 : 0;  // clamped: last iter is a
                                                 // harmless dummy prefetch
    // Stage next B k-slice into the other buffer (async DMA, ASYNCcnt).
    unsigned go = gbase + (unsigned)(kn << 1);
    unsigned ln = cur ? lb0 : lb1;
    asm volatile("global_load_async_to_lds_b128 %0, %1, %2"
                 :: "v"(ln), "v"(go), "s"(Bt) : "memory");
    // Prefetch next A fragments (latency hidden behind this step's WMMAs).
    v16h a0n = load_frag(arow0 + kn);
    v16h a1n = load_frag(arow1 + kn);
    // Four B fragments, 8 WMMAs; ds latency overlaps with WMMA execution.
    const _Float16* lb = &bsh[cur][0];
    v16h b0 = load_frag(lb + (0 * 16 + r) * 32 + klo);
    v16h b1 = load_frag(lb + (1 * 16 + r) * 32 + klo);
    v16h b2 = load_frag(lb + (2 * 16 + r) * 32 + klo);
    v16h b3 = load_frag(lb + (3 * 16 + r) * 32 + klo);
    acc[0][0] = wmma_f16(a0, b0, acc[0][0]);
    acc[1][0] = wmma_f16(a1, b0, acc[1][0]);
    acc[0][1] = wmma_f16(a0, b1, acc[0][1]);
    acc[1][1] = wmma_f16(a1, b1, acc[1][1]);
    acc[0][2] = wmma_f16(a0, b2, acc[0][2]);
    acc[1][2] = wmma_f16(a1, b2, acc[1][2]);
    acc[0][3] = wmma_f16(a0, b3, acc[0][3]);
    acc[1][3] = wmma_f16(a1, b3, acc[1][3]);
    a0 = a0n;
    a1 = a1n;
    async_stage_barrier();  // next buffer staged for everyone; cur reusable
  }
}

// ------------------------------------------------------------- QKV GEMM -----
// qkv = xh[8192,1024] @ W_attn + b_attn ; scatter into qh/kh [B,H,T,HD] and
// vt [B,H,HD,T] (V transposed so PV B-fragments are contiguous).
__global__ __launch_bounds__(256) void gemm_qkv_kernel(
    const _Float16* __restrict__ Ah, const _Float16* __restrict__ Bt,
    const float* __restrict__ bias, _Float16* __restrict__ qh,
    _Float16* __restrict__ kh, _Float16* __restrict__ vt) {
  __shared__ __align__(16) _Float16 bsh[2][64 * 32];
  const int Mblocks = (Bq * Tq) / 256;  // 32
  int wave = threadIdx.x >> 5, lane = threadIdx.x & 31;
  int mb = blockIdx.x % Mblocks, ng = blockIdx.x / Mblocks;
  int m0w = mb * 256 + wave * 32, n0 = ng * 64;
  int r = lane & 15, half = lane >> 4;

  v8f acc[2][4] = {};
  gemm_tile_256x64(Ah, Bt, bsh, m0w, n0, acc);

  // Epilogue: bias + scatter. C layout: lane holds col n=r, rows +g+half*8.
#pragma unroll
  for (int mt = 0; mt < 2; ++mt) {
#pragma unroll
    for (int nt = 0; nt < 4; ++nt) {
      int n = n0 + nt * 16 + r;          // [0, 3072)
      int which = n >> 10;               // 0=q 1=k 2=v
      int e = n & (Eq - 1);
      int h = e >> 6, d = e & (HDq - 1);
      float bv = bias[n];
      if (which == 2) {
        // vt rows are consecutive t for g=0..7 -> one 16B packed store.
        int mbase = m0w + mt * 16 + (half << 3);
        int b = mbase >> 11, t0 = mbase & (Tq - 1);
        v8h pack;
#pragma unroll
        for (int g = 0; g < 8; ++g) pack[g] = (_Float16)(acc[mt][nt][g] + bv);
        *(v8h*)&vt[(((size_t)(b * Hq + h) * HDq + d) << 11) + t0] = pack;
      } else {
#pragma unroll
        for (int g = 0; g < 8; ++g) {
          int m = m0w + mt * 16 + g + (half << 3);
          int b = m >> 11, t = m & (Tq - 1);
          _Float16 val = (_Float16)(acc[mt][nt][g] + bv);
          size_t idx = (((size_t)(b * Hq + h) * Tq + t) << 6) + d;
          if (which == 0) qh[idx] = val; else kh[idx] = val;
        }
      }
    }
  }
}

// -------------------------------------------------------- flash attention ---
// Block = 4 waves; each wave owns 16 query rows, streams causal keys in
// tiles of 32 with online softmax. grid = (T/64, H, B).
__global__ __launch_bounds__(128) void flash_attn_kernel(
    const _Float16* __restrict__ qh, const _Float16* __restrict__ kh,
    const _Float16* __restrict__ vt, _Float16* __restrict__ yh) {
  __shared__ __align__(16) _Float16 plds[4][16 * 32];  // wave-private P tiles

  int wave = threadIdx.x >> 5, lane = threadIdx.x & 31;
  int r = lane & 15, half = lane >> 4, klo = half << 3;
  int h = blockIdx.y, b = blockIdx.z;
  int q0 = blockIdx.x * 64 + wave * 16;

  const _Float16* qp = qh + ((size_t)(b * Hq + h) << 17);  // *T*HD
  const _Float16* kp = kh + ((size_t)(b * Hq + h) << 17);
  const _Float16* vp = vt + ((size_t)(b * Hq + h) << 17);

  // Q fragments for k-dim 0..31 and 32..63, pre-scaled by 1/sqrt(HD)=0.125.
  v16h aq0 = load_frag(qp + (size_t)(q0 + r) * HDq + klo);
  v16h aq1 = load_frag(qp + (size_t)(q0 + r) * HDq + 32 + klo);
#pragma unroll
  for (int i = 0; i < 16; ++i) {
    aq0[i] = aq0[i] * (_Float16)0.125f;
    aq1[i] = aq1[i] * (_Float16)0.125f;
  }

  v8f acc[4] = {};
  float mrow[8], lrow[8];
#pragma unroll
  for (int g = 0; g < 8; ++g) { mrow[g] = NEG_INF; lrow[g] = 0.0f; }

  _Float16* pw = &plds[wave][0];
  int kend = q0 + 15;  // causal: last key this wave can see

  for (int kt = 0; kt <= kend; kt += 32) {
    // S tiles (queries x keys): two 16x16 tiles over a 32-key strip.
    // Distinct K fragments -> clause of 8 global_load_b128.
    v16h bk0 = load_frag(kp + (size_t)(kt + r) * HDq + klo);
    v16h bk1 = load_frag(kp + (size_t)(kt + r) * HDq + 32 + klo);
    v16h bk2 = load_frag(kp + (size_t)(kt + 16 + r) * HDq + klo);
    v16h bk3 = load_frag(kp + (size_t)(kt + 16 + r) * HDq + 32 + klo);
    v8f s0 = {}, s1 = {};
    s0 = wmma_f16(aq0, bk0, s0);
    s0 = wmma_f16(aq1, bk1, s0);
    s1 = wmma_f16(aq0, bk2, s1);
    s1 = wmma_f16(aq1, bk3, s1);
    // Hoist V fragment loads: their latency hides under the softmax VALU work.
    v16h bv0 = load_frag(vp + (size_t)(0 * 16 + r) * Tq + kt + klo);
    v16h bv1 = load_frag(vp + (size_t)(1 * 16 + r) * Tq + kt + klo);
    v16h bv2 = load_frag(vp + (size_t)(2 * 16 + r) * Tq + kt + klo);
    v16h bv3 = load_frag(vp + (size_t)(3 * 16 + r) * Tq + kt + klo);

    int kc = kt + r;  // this lane's key column (tile 0)
#pragma unroll
    for (int g = 0; g < 8; ++g) {
      int qrow = q0 + g + (half << 3);
      float x0 = (kc <= qrow) ? s0[g] : NEG_INF;
      float x1 = (kc + 16 <= qrow) ? s1[g] : NEG_INF;
      // row max across the 16 lanes of this half-wave
      float tm = fmaxf(x0, x1);
      tm = fmaxf(tm, __shfl_xor(tm, 1, 32));
      tm = fmaxf(tm, __shfl_xor(tm, 2, 32));
      tm = fmaxf(tm, __shfl_xor(tm, 4, 32));
      tm = fmaxf(tm, __shfl_xor(tm, 8, 32));
      float mnew = fmaxf(mrow[g], tm);
      float corr = __expf(mrow[g] - mnew);
      mrow[g] = mnew;
      float p0 = __expf(x0 - mnew);
      float p1 = __expf(x1 - mnew);
      float rs = p0 + p1;
      rs += __shfl_xor(rs, 1, 32);
      rs += __shfl_xor(rs, 2, 32);
      rs += __shfl_xor(rs, 4, 32);
      rs += __shfl_xor(rs, 8, 32);
      lrow[g] = lrow[g] * corr + rs;
#pragma unroll
      for (int dt = 0; dt < 4; ++dt) acc[dt][g] *= corr;
      s0[g] = p0;
      s1[g] = p1;
    }
    // Spill P to wave-private LDS (row-major 16x32 f16), reload as A-frag.
#pragma unroll
    for (int g = 0; g < 8; ++g) {
      int row = g + (half << 3);
      pw[row * 32 + r] = (_Float16)s0[g];
      pw[row * 32 + 16 + r] = (_Float16)s1[g];
    }
    asm volatile("s_wait_dscnt 0" ::: "memory");
    v16h ap = load_frag(pw + r * 32 + klo);
    // acc += P (16x32) @ V (32x16) for 4 d-tiles (V fragments preloaded).
    acc[0] = wmma_f16(ap, bv0, acc[0]);
    acc[1] = wmma_f16(ap, bv1, acc[1]);
    acc[2] = wmma_f16(ap, bv2, acc[2]);
    acc[3] = wmma_f16(ap, bv3, acc[3]);
  }

  // Normalize and write y back as [B,T,E] f16 (E col = h*64 + dt*16 + r).
#pragma unroll
  for (int g = 0; g < 8; ++g) {
    float inv = 1.0f / lrow[g];
    int m = (b << 11) + q0 + g + (half << 3);  // b*T + t
#pragma unroll
    for (int dt = 0; dt < 4; ++dt) {
      int e = (h << 6) + dt * 16 + r;
      yh[(size_t)m * Eq + e] = (_Float16)(acc[dt][g] * inv);
    }
  }
}

// ------------------------------------------------------------ proj GEMM -----
__global__ __launch_bounds__(256) void gemm_proj_kernel(
    const _Float16* __restrict__ Ah, const _Float16* __restrict__ Bt,
    const float* __restrict__ bias, float* __restrict__ out) {
  __shared__ __align__(16) _Float16 bsh[2][64 * 32];
  const int Mblocks = (Bq * Tq) / 256;  // 32
  int wave = threadIdx.x >> 5, lane = threadIdx.x & 31;
  int mb = blockIdx.x % Mblocks, ng = blockIdx.x / Mblocks;
  int m0w = mb * 256 + wave * 32, n0 = ng * 64;
  int r = lane & 15, half = lane >> 4;

  v8f acc[2][4] = {};
  gemm_tile_256x64(Ah, Bt, bsh, m0w, n0, acc);

#pragma unroll
  for (int mt = 0; mt < 2; ++mt) {
#pragma unroll
    for (int nt = 0; nt < 4; ++nt) {
      int n = n0 + nt * 16 + r;
      float bv = bias[n];
#pragma unroll
      for (int g = 0; g < 8; ++g) {
        int m = m0w + mt * 16 + g + (half << 3);
        out[(size_t)m * Eq + n] = acc[mt][nt][g] + bv;
      }
    }
  }
}

// ---------------------------------------------------------------- launch ----
extern "C" void kernel_launch(void* const* d_in, const int* in_sizes, int n_in,
                              void* d_out, int out_size, void* d_ws,
                              size_t ws_size, hipStream_t stream) {
  const float* x      = (const float*)d_in[0];  // [B,T,E]
  const float* W_attn = (const float*)d_in[1];  // [E,3E]
  const float* b_attn = (const float*)d_in[2];  // [3E]
  const float* W_proj = (const float*)d_in[3];  // [E,E]
  const float* b_proj = (const float*)d_in[4];  // [E]
  float* out = (float*)d_out;                   // [B,T,E] f32

  // Workspace layout (all 16B-aligned chunk sizes), ~88 MB total.
  char* ws = (char*)d_ws;
  const size_t sz_xh  = (size_t)Bq * Tq * Eq * 2;        // 16 MB
  const size_t sz_wat = (size_t)3 * Eq * Eq * 2;         // 6 MB
  const size_t sz_wpt = (size_t)Eq * Eq * 2;             // 2 MB
  const size_t sz_qkv = (size_t)Bq * Hq * Tq * HDq * 2;  // 16 MB each
  _Float16* xh  = (_Float16*)ws;            ws += sz_xh;
  _Float16* wat = (_Float16*)ws;            ws += sz_wat;
  _Float16* wpt = (_Float16*)ws;            ws += sz_wpt;
  _Float16* qh  = (_Float16*)ws;            ws += sz_qkv;
  _Float16* kh  = (_Float16*)ws;            ws += sz_qkv;
  _Float16* vt  = (_Float16*)ws;            ws += sz_qkv;  // [B,H,HD,T]
  _Float16* yh  = (_Float16*)ws;            ws += sz_xh;   // [B,T,E]

  const int nX = Bq * Tq * Eq;          // 8388608
  const int nWA = 3 * Eq * Eq;          // 3145728
  const int nWP = Eq * Eq;              // 1048576

  f32_to_f16_kernel<<<(nX + 255) / 256, 256, 0, stream>>>(x, xh, nX);
  transpose_to_f16_kernel<<<(nWA + 255) / 256, 256, 0, stream>>>(W_attn, wat,
                                                                 Eq, 3 * Eq);
  transpose_to_f16_kernel<<<(nWP + 255) / 256, 256, 0, stream>>>(W_proj, wpt,
                                                                 Eq, Eq);

  // QKV GEMM: (8192/256) x (3072/64) = 32 x 48 = 1536 blocks.
  gemm_qkv_kernel<<<1536, 256, 0, stream>>>(xh, wat, b_attn, qh, kh, vt);

  // Flash attention: grid (T/64, H, B), 4 waves/block.
  flash_attn_kernel<<<dim3(Tq / 64, Hq, Bq), 128, 0, stream>>>(qh, kh, vt, yh);

  // Output projection: (8192/256) x (1024/64) = 32 x 16 = 512 blocks.
  gemm_proj_kernel<<<512, 256, 0, stream>>>(yh, wpt, b_proj, out);
}
